// GINAttnNet_62818191671501
// MI455X (gfx1250) — matrix operand order repaired and checked by reading, not compiled, not run on hardware
//
#include <hip/hip_runtime.h>
#include <hip/hip_bf16.h>
#include <math.h>

// ---------------------------------------------------------------------------
// GINAttnNet for MI455X (gfx1250, wave32).
// GEMMs: bf16 WMMA (v_wmma_f32_16x16x32_bf16), fp32 accumulate.
//   - CT column-tile accumulators per wave: one A fragment feeds CT WMMAs.
//   - explicit double-buffered K loop: next-step A/B loads issued before the
//     current WMMA group, so global-load latency overlaps the XDL pipe.
//   - strength-reduced B addressing (pointer += 32*M, no per-step mul).
// Edge softmax / aggregation: fp32 with L2-resident global atomics.
// ---------------------------------------------------------------------------

typedef __bf16 bf16_t;
typedef bf16_t v16bf __attribute__((ext_vector_type(16)));
typedef bf16_t v8bf  __attribute__((ext_vector_type(8)));
typedef float  v8f   __attribute__((ext_vector_type(8)));

#define NNODES 50000
#define NEDGES 400000

static inline int cdiv_i(int a, int b) { return (a + b - 1) / b; }

// ---- fp32 -> bf16 conversion with optional zero column-padding -------------
__global__ void k_cvt_pad(const float* __restrict__ in, bf16_t* __restrict__ out,
                          int rows, int K, int Kp) {
  long long i = (long long)blockIdx.x * blockDim.x + threadIdx.x;
  long long total = (long long)rows * Kp;
  if (i >= total) return;
  int r = (int)(i / Kp);
  int c = (int)(i % Kp);
  out[i] = (c < K) ? (bf16_t)in[(long long)r * K + c] : (bf16_t)0.0f;
}

// ---- WMMA GEMM: C[rows x M] = A[rows x K](bf16) * B[K x M](bf16) + bias ----
// One wave per 16 x (16*CT) C strip; 8 waves per block stacked along rows so
// the B strip is shared via L0/L2. K % 32 == 0, M % (16*CT) == 0, rows % 16 == 0.
template <int CT>
__global__ void k_gemm_wmma(const bf16_t* __restrict__ A,
                            const bf16_t* __restrict__ B,
                            const float*  __restrict__ bias,
                            float* __restrict__ C,
                            int rows, int K, int M, int relu) {
  const int lane = threadIdx.x & 31;
  const int wave = threadIdx.x >> 5;
  const int rowTile = blockIdx.y * 8 + wave;
  if (rowTile * 16 >= rows) return;           // wave-uniform guard (EXEC all-1 for WMMA)
  const int col0 = blockIdx.x * (16 * CT);

  // A-fragment lane mapping (ISA 16-bit A 16x32 layout):
  //   row = lane&15 ; lanes<16 hold K-chunks [0..7],[16..23], lanes>=16 [8..15],[24..31]
  const int r  = lane & 15;
  const int kb = (lane >> 4) << 3;            // 0 or 8
  const bf16_t* Ap = A + (size_t)(rowTile * 16 + r) * K;
  // B-fragment: lane holds column N = col0+ct*16+(lane&15), K in the same order as A
  const bf16_t* Bp = B + col0 + r;
  const size_t bstep = (size_t)32 * M;

  union AF { v16bf v; v8bf h[2]; };

  v8f acc[CT];
#pragma unroll
  for (int ct = 0; ct < CT; ++ct) acc[ct] = (v8f){};

  // ---- prologue: load K-step 0 ----
  AF a_cur;
  a_cur.h[0] = *(const v8bf*)(Ap + kb);
  a_cur.h[1] = *(const v8bf*)(Ap + 16 + kb);
  v16bf b_cur[CT];
#pragma unroll
  for (int ct = 0; ct < CT; ++ct)
    for (int s = 0; s < 16; ++s) b_cur[ct][s] = Bp[(size_t)(kb + s + (s & 8)) * M + ct * 16];

  // ---- steady state: prefetch step k0, compute step k0-32 ----
  for (int k0 = 32; k0 < K; k0 += 32) {
    Bp += bstep;
    AF a_nxt;
    a_nxt.h[0] = *(const v8bf*)(Ap + k0 + kb);
    a_nxt.h[1] = *(const v8bf*)(Ap + k0 + 16 + kb);
    v16bf b_nxt[CT];
#pragma unroll
    for (int ct = 0; ct < CT; ++ct)
      for (int s = 0; s < 16; ++s) b_nxt[ct][s] = Bp[(size_t)(kb + s + (s & 8)) * M + ct * 16];

#pragma unroll
    for (int ct = 0; ct < CT; ++ct)
      acc[ct] = __builtin_amdgcn_wmma_f32_16x16x32_bf16(false, a_cur.v, false, b_cur[ct],
                                                        (short)0, acc[ct], false, false);
    a_cur = a_nxt;
#pragma unroll
    for (int ct = 0; ct < CT; ++ct) b_cur[ct] = b_nxt[ct];
  }

  // ---- epilogue: last K-step ----
#pragma unroll
  for (int ct = 0; ct < CT; ++ct)
    acc[ct] = __builtin_amdgcn_wmma_f32_16x16x32_bf16(false, a_cur.v, false, b_cur[ct],
                                                      (short)0, acc[ct], false, false);

  // C/D layout: VGPR i -> M = (lane<16 ? i : 8+i), N = lane&15
  const int m0 = rowTile * 16 + ((lane < 16) ? 0 : 8);
#pragma unroll
  for (int ct = 0; ct < CT; ++ct) {
    const int cn = col0 + ct * 16 + r;
    const float bn = bias[cn];
#pragma unroll
    for (int i = 0; i < 8; ++i) {
      float v = acc[ct][i] + bn;
      if (relu) v = fmaxf(v, 0.0f);
      C[(size_t)(m0 + i) * M + cn] = v;
    }
  }
}

static void launch_gemm(const bf16_t* A, const bf16_t* B, const float* bias,
                        float* C, int rows, int K, int M, int relu,
                        hipStream_t stream) {
  const int rowBlocks = cdiv_i(rows / 16, 8);
  dim3 blk(256);
  if (M % 64 == 0) {
    dim3 g((unsigned)(M / 64), (unsigned)rowBlocks);
    k_gemm_wmma<4><<<g, blk, 0, stream>>>(A, B, bias, C, rows, K, M, relu);
  } else if (M % 32 == 0) {
    dim3 g((unsigned)(M / 32), (unsigned)rowBlocks);
    k_gemm_wmma<2><<<g, blk, 0, stream>>>(A, B, bias, C, rows, K, M, relu);
  } else {
    dim3 g((unsigned)(M / 16), (unsigned)rowBlocks);
    k_gemm_wmma<1><<<g, blk, 0, stream>>>(A, B, bias, C, rows, K, M, relu);
  }
}

// ---- edge-softmax support ---------------------------------------------------
__device__ inline void atomicMaxF(float* addr, float val) {
  // sign-split trick: monotone int ordering for non-negative floats,
  // reversed unsigned ordering for negative floats. Identity = -inf.
  if (val >= 0.0f) atomicMax((int*)addr, __float_as_int(val));
  else             atomicMin((unsigned int*)addr, __float_as_uint(val));
}

__global__ void k_init_md(float* __restrict__ m, float* __restrict__ denom, int n) {
  int i = blockIdx.x * blockDim.x + threadIdx.x;
  if (i >= n) return;
  m[i] = -INFINITY;
  denom[i] = 0.0f;
}

__global__ void k_scores(const int* __restrict__ src, const int* __restrict__ dst,
                         const float* __restrict__ q, const float* __restrict__ kk,
                         float* __restrict__ sc, float* __restrict__ m,
                         int E, int a, float scale) {
  int e = blockIdx.x * blockDim.x + threadIdx.x;
  if (e >= E) return;
  int s = src[e], d = dst[e];
  const float4* qd = (const float4*)(q  + (size_t)d * a);
  const float4* ks = (const float4*)(kk + (size_t)s * a);
  float acc = 0.0f;
  for (int i = 0; i < (a >> 2); ++i) {
    float4 qa = qd[i], kb = ks[i];
    acc += qa.x * kb.x + qa.y * kb.y + qa.z * kb.z + qa.w * kb.w;
  }
  acc *= scale;
  sc[e] = acc;
  atomicMaxF(&m[d], acc);
}

__global__ void k_expsum(const int* __restrict__ dst, float* __restrict__ sc,
                         const float* __restrict__ m, float* __restrict__ denom, int E) {
  int e = blockIdx.x * blockDim.x + threadIdx.x;
  if (e >= E) return;
  int d = dst[e];
  float v = __expf(sc[e] - m[d]);
  sc[e] = v;                       // sc now holds exp-weights
  atomicAdd(&denom[d], v);
}

// h[dst] += alpha * x[src], 4 features per thread (float4 gather, f32 atomics
// that resolve in the 192MB L2 since h is L2-resident).
__global__ void k_aggregate(const int* __restrict__ src, const int* __restrict__ dst,
                            const float* __restrict__ x, const float* __restrict__ ew,
                            const float* __restrict__ denom, float* __restrict__ h,
                            int E, int din) {
  long long i = (long long)blockIdx.x * blockDim.x + threadIdx.x;
  int chunks = din >> 2;
  long long total = (long long)E * chunks;
  if (i >= total) return;
  int e = (int)(i / chunks);
  int c = (int)(i % chunks) << 2;
  int s = src[e], d = dst[e];
  float alpha = ew[e] / (denom[d] + 1e-16f);
  float4 xs = *(const float4*)(x + (size_t)s * din + c);
  float* hp = h + (size_t)d * din + c;
  atomicAdd(hp + 0, alpha * xs.x);
  atomicAdd(hp + 1, alpha * xs.y);
  atomicAdd(hp + 2, alpha * xs.z);
  atomicAdd(hp + 3, alpha * xs.w);
}

__global__ void k_logsoftmax16(float* __restrict__ z, int n) {
  int i = blockIdx.x * blockDim.x + threadIdx.x;
  if (i >= n) return;
  float* p = z + (size_t)i * 16;
  float v[16];
  float mx = -INFINITY;
#pragma unroll
  for (int j = 0; j < 16; ++j) { v[j] = p[j]; mx = fmaxf(mx, v[j]); }
  float sum = 0.0f;
#pragma unroll
  for (int j = 0; j < 16; ++j) sum += __expf(v[j] - mx);
  float lg = mx + __logf(sum);
#pragma unroll
  for (int j = 0; j < 16; ++j) p[j] = v[j] - lg;
}

// ---------------------------------------------------------------------------
// Host-side per-layer driver (all launches on `stream`).
// ---------------------------------------------------------------------------
static void run_layer(const float* x, int din, int dout, int a,
                      const float* Wq, const float* bq,
                      const float* Wk, const float* bk,
                      const float* W1, const float* b1,
                      const float* W2, const float* b2,
                      float* out, int relu_out, int Kp2,
                      bf16_t* wqb, bf16_t* wkb, bf16_t* w1b, bf16_t* w2b,
                      bf16_t* xb, bf16_t* hb, bf16_t* tb,
                      float* q, float* k, float* sc, float* m, float* dn,
                      float* h,               // also reused as t (MLP hidden)
                      const int* src, const int* dst,
                      hipStream_t stream) {
  const int N = NNODES, E = NEDGES;
  dim3 blk(256);
  auto cvt = [&](const float* in, bf16_t* o, int rows, int K, int Kp) {
    long long tot = (long long)rows * Kp;
    k_cvt_pad<<<(unsigned)((tot + 255) / 256), blk, 0, stream>>>(in, o, rows, K, Kp);
  };

  // bf16 mirrors of weights (W2 zero-padded along K for the K=16 case)
  cvt(Wq, wqb, 1, din * a, din * a);
  cvt(Wk, wkb, 1, din * a, din * a);
  cvt(W1, w1b, 1, din * dout, din * dout);
  cvt(W2, w2b, 1, dout * dout, Kp2 * dout);
  // bf16 mirror of input features
  cvt(x, xb, N, din, din);

  // q / k projections
  launch_gemm(xb, wqb, bq, q, N, din, a, 0, stream);
  launch_gemm(xb, wkb, bk, k, N, din, a, 0, stream);

  // segment softmax over edges grouped by dst
  k_init_md<<<cdiv_i(N, 256), blk, 0, stream>>>(m, dn, N);
  k_scores<<<cdiv_i(E, 256), blk, 0, stream>>>(src, dst, q, k, sc, m, E, a,
                                               1.0f / sqrtf((float)a));
  k_expsum<<<cdiv_i(E, 256), blk, 0, stream>>>(dst, sc, m, dn, E);

  // h = x + sum_j alpha_ij x_j   (init h=x, then scatter-add)
  hipMemcpyAsync(h, x, (size_t)N * din * sizeof(float),
                 hipMemcpyDeviceToDevice, stream);
  long long aggTot = (long long)E * (din / 4);
  k_aggregate<<<(unsigned)((aggTot + 255) / 256), blk, 0, stream>>>(
      src, dst, x, sc, dn, h, E, din);

  // MLP: t = relu(h @ W1 + b1);  out = t @ W2 + b2 (+outer relu)
  cvt(h, hb, N, din, din);
  float* t = h;  // h is dead once hb exists; reuse its storage for t
  launch_gemm(hb, w1b, b1, t, N, din, dout, 1, stream);
  cvt(t, tb, N, dout, Kp2);
  launch_gemm(tb, w2b, b2, out, N, Kp2, dout, relu_out, stream);
}

extern "C" void kernel_launch(void* const* d_in, const int* in_sizes, int n_in,
                              void* d_out, int out_size, void* d_ws, size_t ws_size,
                              hipStream_t stream) {
  const float* x   = (const float*)d_in[0];
  const int*   ei  = (const int*)d_in[1];
  const int*   src = ei;            // edge_index[0]
  const int*   dst = ei + NEDGES;   // edge_index[1]

  const float* Wq1 = (const float*)d_in[2];  const float* bq1 = (const float*)d_in[3];
  const float* Wk1 = (const float*)d_in[4];  const float* bk1 = (const float*)d_in[5];
  const float* W11 = (const float*)d_in[6];  const float* b11 = (const float*)d_in[7];
  const float* W21 = (const float*)d_in[8];  const float* b21 = (const float*)d_in[9];
  const float* Wq2 = (const float*)d_in[10]; const float* bq2 = (const float*)d_in[11];
  const float* Wk2 = (const float*)d_in[12]; const float* bk2 = (const float*)d_in[13];
  const float* W12 = (const float*)d_in[14]; const float* b12 = (const float*)d_in[15];
  const float* W22 = (const float*)d_in[16]; const float* b22 = (const float*)d_in[17];
  const float* Wq3 = (const float*)d_in[18]; const float* bq3 = (const float*)d_in[19];
  const float* Wk3 = (const float*)d_in[20]; const float* bk3 = (const float*)d_in[21];
  const float* W13 = (const float*)d_in[22]; const float* b13 = (const float*)d_in[23];
  const float* W23 = (const float*)d_in[24]; const float* b23 = (const float*)d_in[25];

  // carve workspace (256B aligned slices)
  char* p = (char*)d_ws;
  auto alloc = [&](size_t bytes) -> void* {
    void* r = (void*)p;
    p += (bytes + 255) & ~(size_t)255;
    return r;
  };
  bf16_t* wqb = (bf16_t*)alloc((size_t)512 * 64 * 2);
  bf16_t* wkb = (bf16_t*)alloc((size_t)512 * 64 * 2);
  bf16_t* w1b = (bf16_t*)alloc((size_t)512 * 512 * 2);
  bf16_t* w2b = (bf16_t*)alloc((size_t)512 * 512 * 2);
  bf16_t* xb  = (bf16_t*)alloc((size_t)NNODES * 512 * 2);
  bf16_t* hb  = (bf16_t*)alloc((size_t)NNODES * 512 * 2);
  bf16_t* tb  = (bf16_t*)alloc((size_t)NNODES * 512 * 2);
  float*  q   = (float*)alloc((size_t)NNODES * 64 * 4);
  float*  k   = (float*)alloc((size_t)NNODES * 64 * 4);
  float*  sc  = (float*)alloc((size_t)NEDGES * 4);
  float*  m   = (float*)alloc((size_t)NNODES * 4);
  float*  dn  = (float*)alloc((size_t)NNODES * 4);
  float*  h   = (float*)alloc((size_t)NNODES * 512 * 4);   // also MLP hidden t
  float*  F1  = (float*)alloc((size_t)NNODES * 512 * 4);   // layer-1 output
  float*  F2  = (float*)alloc((size_t)NNODES * 256 * 4);   // layer-2 output
  (void)in_sizes; (void)n_in; (void)out_size; (void)ws_size;

  // layer 1: 256 -> 512 (a=64), outer relu
  run_layer(x,  256, 512, 64, Wq1, bq1, Wk1, bk1, W11, b11, W21, b21,
            F1, 1, 512, wqb, wkb, w1b, w2b, xb, hb, tb,
            q, k, sc, m, dn, h, src, dst, stream);
  // layer 2: 512 -> 256 (a=64), outer relu
  run_layer(F1, 512, 256, 64, Wq2, bq2, Wk2, bk2, W12, b12, W22, b22,
            F2, 1, 256, wqb, wkb, w1b, w2b, xb, hb, tb,
            q, k, sc, m, dn, h, src, dst, stream);
  // layer 3: 256 -> 16 (a=32), no relu; K of second GEMM padded 16 -> 32
  run_layer(F2, 256, 16, 32, Wq3, bq3, Wk3, bk3, W13, b13, W23, b23,
            (float*)d_out, 0, 32, wqb, wkb, w1b, w2b, xb, hb, tb,
            q, k, sc, m, dn, h, src, dst, stream);

  // final log_softmax over C=16, in place on d_out
  k_logsoftmax16<<<cdiv_i(NNODES, 256), 256, 0, stream>>>((float*)d_out, NNODES);
}
